// Adapter_XFormersAttnProcessor_20873541058986
// MI455X (gfx1250) — matrix operand order repaired and checked by reading, not compiled, not run on hardware
//
#include <hip/hip_runtime.h>
#include <hip/hip_bf16.h>

// ---------------------------------------------------------------------------
// Problem constants
// ---------------------------------------------------------------------------
#define BATCH   16
#define QLEN    1024
#define CDIM    1280
#define CROSS   2048
#define LTOT    477
#define HEADS   20
#define HDIM    64
#define LKV1    96     // 77 padded to 96 (3 K-chunks of 32)
#define NV1     77
#define LKV2    224    // 200 padded to 224 (7 K-chunks of 32)
#define NV2     200
#define ATTN_SCALE 0.125f   // 1/sqrt(64)

typedef __bf16 bf16x8  __attribute__((ext_vector_type(8)));
typedef __bf16 bf16x16 __attribute__((ext_vector_type(16)));
typedef float  v8f     __attribute__((ext_vector_type(8)));

// ---------------------------------------------------------------------------
// WMMA helpers (CDNA5: V_WMMA_F32_16X16X32_BF16, fp32 accumulate)
// ---------------------------------------------------------------------------
__device__ __forceinline__ v8f wmma_bf16(bf16x16 a, bf16x16 b, v8f c) {
  return __builtin_amdgcn_wmma_f32_16x16x32_bf16(
      /*neg_a=*/false, a, /*neg_b=*/false, b,
      /*c_mod=*/(short)0, c, /*reuse_a=*/false, /*reuse_b=*/false);
}

// A-fragment per-lane layout (16-bit, 16x32): lane holds m=(l&15),
// K values: base..base+7 (VGPR0-3) and base+16..base+23 (VGPR4-7), base=(l>>4)*8.
// Caller passes p = row + base: lo at p[0..7], hi at p[16..23].
__device__ __forceinline__ bf16x16 load_frag_a(const __bf16* p) {
  bf16x8 lo = *(const bf16x8*)p;
  bf16x8 hi = *(const bf16x8*)(p + 16);
  return __builtin_shufflevector(lo, hi, 0,1,2,3,4,5,6,7,8,9,10,11,12,13,14,15);
}

// B-fragment per-lane layout (16-bit, 32x16): lane holds n=(l&15),
// K values: base..base+7 and base+16..base+23, base=(l>>4)*8. Caller passes p = row + (l>>4)*8.
__device__ __forceinline__ bf16x16 load_frag_b(const __bf16* p) {
  bf16x8 lo = *(const bf16x8*)p;
  bf16x8 hi = *(const bf16x8*)(p + 16);
  return __builtin_shufflevector(lo, hi, 0,1,2,3,4,5,6,7,8,9,10,11,12,13,14,15);
}

// ---------------------------------------------------------------------------
// Projection GEMM: Out = X(fp32) @ W(fp32,(N,K) row-major).T  -> bf16
// head-split output: transpose_out=0 -> [b][h][j][d], =1 -> [b][h][d][j]
// Block = 128 threads (4 waves), 64x64 tile, K-step 32 staged in LDS as bf16.
// ---------------------------------------------------------------------------
__global__ __launch_bounds__(128) void proj_gemm(
    const float* __restrict__ X, const float* __restrict__ W,
    __bf16* __restrict__ Out, int K, int rows_src, int rows_pad,
    int src_rows_total, int src_row_off, int transpose_out)
{
  __shared__ __bf16 As[64][32];
  __shared__ __bf16 Bs[64][32];
  const int tid  = threadIdx.x;
  const int lane = tid & 31;
  const int wv   = tid >> 5;
  const int wm   = wv >> 1, wn = wv & 1;
  const int srow = tid >> 1;
  const int scol = (tid & 1) * 16;
  const int l15  = lane & 15;
  const int abase = (lane >> 4) * 8;
  const int bbase = (lane >> 4) * 8;

  const int rg = blockIdx.x * 64 + srow;
  const int bb = rg / rows_pad, j = rg % rows_pad;
  const bool avalid = (j < rows_src);
  const float* arow = X + ((size_t)bb * src_rows_total + src_row_off + j) * K;
  const float* brow = W + ((size_t)(blockIdx.y * 64 + srow)) * K;

  v8f acc[2][2] = {};

  for (int kk = 0; kk < K; kk += 32) {
    // ---- stage A panel (64x32 fp32 -> bf16) ----
    if (avalid) {
      const float4* ap = (const float4*)(arow + kk + scol);
      __builtin_prefetch(arow + kk + 256, 0, 1);
#pragma unroll
      for (int q = 0; q < 4; ++q) {
        float4 v = ap[q];
        As[srow][scol + q * 4 + 0] = (__bf16)v.x;
        As[srow][scol + q * 4 + 1] = (__bf16)v.y;
        As[srow][scol + q * 4 + 2] = (__bf16)v.z;
        As[srow][scol + q * 4 + 3] = (__bf16)v.w;
      }
    } else {
#pragma unroll
      for (int q = 0; q < 16; ++q) As[srow][scol + q] = (__bf16)0.0f;
    }
    // ---- stage B panel (W rows are output features, contiguous K) ----
    {
      const float4* bp = (const float4*)(brow + kk + scol);
      __builtin_prefetch(brow + kk + 256, 0, 1);
#pragma unroll
      for (int q = 0; q < 4; ++q) {
        float4 v = bp[q];
        Bs[srow][scol + q * 4 + 0] = (__bf16)v.x;
        Bs[srow][scol + q * 4 + 1] = (__bf16)v.y;
        Bs[srow][scol + q * 4 + 2] = (__bf16)v.z;
        Bs[srow][scol + q * 4 + 3] = (__bf16)v.w;
      }
    }
    __syncthreads();
#pragma unroll
    for (int mt = 0; mt < 2; ++mt) {
      bf16x16 af = load_frag_a(&As[wm * 32 + mt * 16 + l15][abase]);
#pragma unroll
      for (int nt = 0; nt < 2; ++nt) {
        bf16x16 bf_ = load_frag_b(&Bs[wn * 32 + nt * 16 + l15][bbase]);
        acc[mt][nt] = wmma_bf16(af, bf_, acc[mt][nt]);
      }
    }
    __syncthreads();
  }

  // ---- epilogue: scatter C fragments into head-split layout ----
#pragma unroll
  for (int mt = 0; mt < 2; ++mt)
#pragma unroll
    for (int nt = 0; nt < 2; ++nt)
#pragma unroll
      for (int r = 0; r < 8; ++r) {
        int mloc = wm * 32 + mt * 16 + r + (lane >> 4) * 8;
        int nloc = wn * 32 + nt * 16 + l15;
        int rg2  = blockIdx.x * 64 + mloc;
        int b2   = rg2 / rows_pad, j2 = rg2 % rows_pad;
        int ng   = blockIdx.y * 64 + nloc;
        int h    = ng >> 6, d = ng & 63;
        __bf16 val = (__bf16)acc[mt][nt][r];
        if (!transpose_out)
          Out[(((size_t)b2 * HEADS + h) * rows_pad + j2) * HDIM + d] = val;
        else
          Out[(((size_t)b2 * HEADS + h) * HDIM + d) * rows_pad + j2] = val;
      }
}

// ---------------------------------------------------------------------------
// One softmax-attention pass for a 16-row query strip (per wave, no barriers:
// each wave owns a private LDS S/P strip). Accumulates O into acc[4] (16x64).
// kmat: [LKV][64] row-major, vT: [64][LKV] transposed.
// ---------------------------------------------------------------------------
__device__ __forceinline__ void attn_pass(
    const __bf16* __restrict__ kmat, const __bf16* __restrict__ vT,
    float* S, __bf16* P, bf16x16 qa0, bf16x16 qa1,
    int LKV, int nvalid, int lane, v8f acc[4])
{
  const int l15 = lane & 15;
  const int abase = (lane >> 4) * 8;
  const int bbase = (lane >> 4) * 8;

  // ---- S = Q K^T (per 16x16 key tile, 2 WMMAs over d=64) ----
  for (int j0 = 0; j0 < LKV; j0 += 16) {
    const __bf16* kr = kmat + (size_t)(j0 + l15) * HDIM;
    bf16x16 b0 = load_frag_b(kr + bbase);
    bf16x16 b1 = load_frag_b(kr + 32 + bbase);
    v8f s = {};
    s = wmma_bf16(qa0, b0, s);
    s = wmma_bf16(qa1, b1, s);
    int col = j0 + l15;
#pragma unroll
    for (int r = 0; r < 8; ++r) S[(abase + r) * LKV + col] = s[r];
  }
  asm volatile("s_wait_dscnt 0" ::: "memory");

  // ---- softmax over each row (2 lanes per row, shfl_xor(16) combine) ----
  const int row  = l15;
  const int half = lane >> 4;
  const int c0 = half * (LKV >> 1), c1 = c0 + (LKV >> 1);
  float*  Sr = S + row * LKV;
  __bf16* Pr = P + row * LKV;
  float mx = -1e30f;
  for (int j = c0; j < c1; ++j)
    if (j < nvalid) mx = fmaxf(mx, Sr[j] * ATTN_SCALE);
  mx = fmaxf(mx, __shfl_xor(mx, 16, 32));
  float sum = 0.0f;
  for (int j = c0; j < c1; ++j) {
    float e = (j < nvalid) ? __expf(Sr[j] * ATTN_SCALE - mx) : 0.0f;
    sum += e;
    Pr[j] = (__bf16)e;
  }
  sum += __shfl_xor(sum, 16, 32);
  float inv = 1.0f / sum;
  for (int j = c0; j < c1; ++j) Pr[j] = (__bf16)((float)Pr[j] * inv);
  asm volatile("s_wait_dscnt 0" ::: "memory");

  // ---- O += P V  (P as A-fragments from LDS, V^T as B-fragments) ----
  for (int k0 = 0; k0 < LKV; k0 += 32) {
    bf16x16 pa = load_frag_a(P + (size_t)l15 * LKV + k0 + abase);
#pragma unroll
    for (int nt = 0; nt < 4; ++nt) {
      const __bf16* vr = vT + (size_t)(nt * 16 + l15) * LKV + k0 + bbase;
      bf16x16 bv_ = load_frag_b(vr);
      acc[nt] = wmma_bf16(pa, bv_, acc[nt]);
    }
  }
}

// ---------------------------------------------------------------------------
// Box key/value self-update: bk2 = bk + MEA(bk, ck, cv) (row-major out),
//                            bv2T = (bv + MEA(bv, ck, cv))^T
// grid = (B*H, 2[mode]), block = 128 (4 waves), 14 query strips per (b,h).
// ---------------------------------------------------------------------------
__global__ __launch_bounds__(128) void box_update_kernel(
    const __bf16* __restrict__ bk, const __bf16* __restrict__ bv,
    const __bf16* __restrict__ ck, const __bf16* __restrict__ cvT,
    __bf16* __restrict__ bk2, __bf16* __restrict__ bv2T)
{
  extern __shared__ char smem[];
  const int tid = threadIdx.x, lane = tid & 31, wv = tid >> 5;
  float*  S = (float*)smem + (size_t)wv * 16 * LKV2;
  __bf16* P = (__bf16*)(smem + 4 * 16 * LKV2 * 4) + (size_t)wv * 16 * LKV2;
  const int bh   = blockIdx.x;
  const int mode = blockIdx.y;               // 0: bk update, 1: bv update
  const size_t base = (size_t)bh * LKV2 * HDIM;  // same elems for [j][d]/[d][j]
  const __bf16* qsrc = mode ? bv : bk;
  const int l15 = lane & 15, abase = (lane >> 4) * 8;

  for (int s = wv; s < LKV2 / 16; s += 4) {
    const __bf16* qrow = qsrc + base + (size_t)(s * 16 + l15) * HDIM;
    bf16x16 qa0 = load_frag_a(qrow + abase);
    bf16x16 qa1 = load_frag_a(qrow + 32 + abase);
    v8f acc[4] = {};
    attn_pass(ck + base, cvT + base, S, P, qa0, qa1, LKV2, NV2, lane, acc);
#pragma unroll
    for (int nt = 0; nt < 4; ++nt)
#pragma unroll
      for (int r = 0; r < 8; ++r) {
        int n = nt * 16 + l15;
        int rowq = s * 16 + r + (lane >> 4) * 8;
        float orig = (float)qsrc[base + (size_t)rowq * HDIM + n];
        float val = orig + acc[nt][r];
        if (mode == 0)
          bk2[base + (size_t)rowq * HDIM + n] = (__bf16)val;
        else
          bv2T[base + (size_t)n * LKV2 + rowq] = (__bf16)val;
      }
  }
}

// ---------------------------------------------------------------------------
// Fused main attention: accum = MEA(q,k,v) + MEA(q,bk2,bv2) (bf16 out, B,Q,C)
// grid = (Q/64, B*H), block = 128 (4 waves x 16-query strips).
// ---------------------------------------------------------------------------
__global__ __launch_bounds__(128) void attn_main_kernel(
    const __bf16* __restrict__ qh,  const __bf16* __restrict__ kh,
    const __bf16* __restrict__ vhT, const __bf16* __restrict__ bk2,
    const __bf16* __restrict__ bv2T, __bf16* __restrict__ accum)
{
  extern __shared__ char smem[];
  const int tid = threadIdx.x, lane = tid & 31, wv = tid >> 5;
  float*  S = (float*)smem + (size_t)wv * 16 * LKV2;
  __bf16* P = (__bf16*)(smem + 4 * 16 * LKV2 * 4) + (size_t)wv * 16 * LKV2;
  const int bh = blockIdx.y, b = bh / HEADS, h = bh % HEADS;
  const int q0 = blockIdx.x * 64 + wv * 16;
  const int l15 = lane & 15, abase = (lane >> 4) * 8;

  const __bf16* qrow = qh + ((size_t)bh * QLEN + q0 + l15) * HDIM;
  bf16x16 qa0 = load_frag_a(qrow + abase);
  bf16x16 qa1 = load_frag_a(qrow + 32 + abase);

  v8f acc[4] = {};
  const size_t kv1 = (size_t)bh * LKV1 * HDIM;
  attn_pass(kh + kv1, vhT + kv1, S, P, qa0, qa1, LKV1, NV1, lane, acc);
  const size_t kv2 = (size_t)bh * LKV2 * HDIM;
  attn_pass(bk2 + kv2, bv2T + kv2, S, P, qa0, qa1, LKV2, NV2, lane, acc);

#pragma unroll
  for (int nt = 0; nt < 4; ++nt)
#pragma unroll
    for (int r = 0; r < 8; ++r) {
      int n = h * HDIM + nt * 16 + l15;
      int rowq = q0 + r + (lane >> 4) * 8;
      accum[((size_t)b * QLEN + rowq) * CDIM + n] = (__bf16)acc[nt][r];
    }
}

// ---------------------------------------------------------------------------
// Output projection: Out(fp32) = accum(bf16) @ w_out.T + b_out
// ---------------------------------------------------------------------------
__global__ __launch_bounds__(128) void out_proj_gemm(
    const __bf16* __restrict__ A, const float* __restrict__ W,
    const float* __restrict__ bias, float* __restrict__ Out)
{
  __shared__ __bf16 As[64][32];
  __shared__ __bf16 Bs[64][32];
  const int tid = threadIdx.x, lane = tid & 31, wv = tid >> 5;
  const int wm = wv >> 1, wn = wv & 1;
  const int srow = tid >> 1, scol = (tid & 1) * 16;
  const int l15 = lane & 15;
  const int abase = (lane >> 4) * 8, bbase = (lane >> 4) * 8;

  const __bf16* arow = A + ((size_t)(blockIdx.x * 64 + srow)) * CDIM;
  const float*  brow = W + ((size_t)(blockIdx.y * 64 + srow)) * CDIM;
  v8f acc[2][2] = {};

  for (int kk = 0; kk < CDIM; kk += 32) {
    *(bf16x8*)&As[srow][scol]     = *(const bf16x8*)(arow + kk + scol);
    *(bf16x8*)&As[srow][scol + 8] = *(const bf16x8*)(arow + kk + scol + 8);
    const float4* bp = (const float4*)(brow + kk + scol);
    __builtin_prefetch(brow + kk + 256, 0, 1);
#pragma unroll
    for (int q = 0; q < 4; ++q) {
      float4 v = bp[q];
      Bs[srow][scol + q * 4 + 0] = (__bf16)v.x;
      Bs[srow][scol + q * 4 + 1] = (__bf16)v.y;
      Bs[srow][scol + q * 4 + 2] = (__bf16)v.z;
      Bs[srow][scol + q * 4 + 3] = (__bf16)v.w;
    }
    __syncthreads();
#pragma unroll
    for (int mt = 0; mt < 2; ++mt) {
      bf16x16 af = load_frag_a(&As[wm * 32 + mt * 16 + l15][abase]);
#pragma unroll
      for (int nt = 0; nt < 2; ++nt) {
        bf16x16 bf_ = load_frag_b(&Bs[wn * 32 + nt * 16 + l15][bbase]);
        acc[mt][nt] = wmma_bf16(af, bf_, acc[mt][nt]);
      }
    }
    __syncthreads();
  }
#pragma unroll
  for (int mt = 0; mt < 2; ++mt)
#pragma unroll
    for (int nt = 0; nt < 2; ++nt)
#pragma unroll
      for (int r = 0; r < 8; ++r) {
        int mloc = wm * 32 + mt * 16 + r + (lane >> 4) * 8;
        int nloc = wn * 32 + nt * 16 + l15;
        size_t rg2 = (size_t)blockIdx.x * 64 + mloc;
        int ng = blockIdx.y * 64 + nloc;
        Out[rg2 * CDIM + ng] = acc[mt][nt][r] + bias[ng];
      }
}

// ---------------------------------------------------------------------------
// Host launcher
// ---------------------------------------------------------------------------
extern "C" void kernel_launch(void* const* d_in, const int* in_sizes, int n_in,
                              void* d_out, int out_size, void* d_ws, size_t ws_size,
                              hipStream_t stream)
{
  (void)in_sizes; (void)n_in; (void)out_size; (void)ws_size;
  const float* hidden  = (const float*)d_in[0];
  const float* enc     = (const float*)d_in[1];
  const float* w_q     = (const float*)d_in[2];
  const float* w_k     = (const float*)d_in[3];
  const float* w_v     = (const float*)d_in[4];
  const float* w_k_box = (const float*)d_in[5];
  const float* w_v_box = (const float*)d_in[6];
  const float* w_k_cls = (const float*)d_in[7];
  const float* w_v_cls = (const float*)d_in[8];
  const float* w_out   = (const float*)d_in[9];
  const float* b_out   = (const float*)d_in[10];
  float* out = (float*)d_out;

  // Workspace carve-up (all bf16, fully written by the producing kernels)
  __bf16* p = (__bf16*)d_ws;
  __bf16* qh    = p; p += (size_t)BATCH * HEADS * QLEN * HDIM;   // q  [b,h,q,d]
  __bf16* kh    = p; p += (size_t)BATCH * HEADS * LKV1 * HDIM;   // k  [b,h,j,d]
  __bf16* vhT   = p; p += (size_t)BATCH * HEADS * HDIM * LKV1;   // v  [b,h,d,j]
  __bf16* bk    = p; p += (size_t)BATCH * HEADS * LKV2 * HDIM;   // bk [b,h,j,d]
  __bf16* bv    = p; p += (size_t)BATCH * HEADS * LKV2 * HDIM;   // bv [b,h,j,d]
  __bf16* ck    = p; p += (size_t)BATCH * HEADS * LKV2 * HDIM;   // ck [b,h,j,d]
  __bf16* cvT   = p; p += (size_t)BATCH * HEADS * HDIM * LKV2;   // cv [b,h,d,j]
  __bf16* bk2   = p; p += (size_t)BATCH * HEADS * LKV2 * HDIM;   // updated bk
  __bf16* bv2T  = p; p += (size_t)BATCH * HEADS * HDIM * LKV2;   // updated bv^T
  __bf16* accum = p; p += (size_t)BATCH * QLEN * CDIM;           // attn out

  dim3 blk(128);
  // Projections (X @ W.T), head-split outputs; V-tensors stored transposed
  proj_gemm<<<dim3(256, 20), blk, 0, stream>>>(hidden, w_q,    qh,  CDIM,  QLEN, QLEN, QLEN, 0,   0);
  proj_gemm<<<dim3( 24, 20), blk, 0, stream>>>(enc,    w_k,    kh,  CROSS, NV1,  LKV1, LTOT, 0,   0);
  proj_gemm<<<dim3( 24, 20), blk, 0, stream>>>(enc,    w_v,    vhT, CROSS, NV1,  LKV1, LTOT, 0,   1);
  proj_gemm<<<dim3( 56, 20), blk, 0, stream>>>(enc,    w_k_box, bk, CROSS, NV2,  LKV2, LTOT, 77,  0);
  proj_gemm<<<dim3( 56, 20), blk, 0, stream>>>(enc,    w_v_box, bv, CROSS, NV2,  LKV2, LTOT, 77,  0);
  proj_gemm<<<dim3( 56, 20), blk, 0, stream>>>(enc,    w_k_cls, ck, CROSS, NV2,  LKV2, LTOT, 277, 0);
  proj_gemm<<<dim3( 56, 20), blk, 0, stream>>>(enc,    w_v_cls, cvT, CROSS, NV2, LKV2, LTOT, 277, 1);

  const size_t lds_attn = 4 * (16 * LKV2 * 4 + 16 * LKV2 * 2);  // 86016 B
  (void)hipFuncSetAttribute((const void*)box_update_kernel,
      hipFuncAttributeMaxDynamicSharedMemorySize, (int)lds_attn);
  (void)hipFuncSetAttribute((const void*)attn_main_kernel,
      hipFuncAttributeMaxDynamicSharedMemorySize, (int)lds_attn);

  // bk/bv self-attention update over cls tokens
  box_update_kernel<<<dim3(BATCH * HEADS, 2), blk, lds_attn, stream>>>(
      bk, bv, ck, cvT, bk2, bv2T);

  // Fused main + box attention, accumulated into one output
  attn_main_kernel<<<dim3(QLEN / 64, BATCH * HEADS), blk, lds_attn, stream>>>(
      qh, kh, vhT, bk2, bv2T, accum);

  // Final projection + bias -> fp32 output
  out_proj_gemm<<<dim3(256, 20), blk, 0, stream>>>(accum, w_out, b_out, out);
}
